// Cycle_Consistency_Loss_61022895341802
// MI455X (gfx1250) — matrix operand order repaired and checked by reading, not compiled, not run on hardware
//
#include <hip/hip_runtime.h>
#include <hip/hip_bf16.h>

typedef __attribute__((ext_vector_type(16))) _Float16 v16h;
typedef __attribute__((ext_vector_type(8)))  _Float16 v8h;
typedef __attribute__((ext_vector_type(8)))  float    v8f;

#define DFEAT 32
#define LOG2E 1.4426950408889634f

__device__ __forceinline__ v8f wmma16x16x32(v16h a, v16h b, v8f c) {
  // D = A(16x32 f16) * B(32x16 f16) + C(16x16 f32)
  return __builtin_amdgcn_wmma_f32_16x16x32_f16(false, a, false, b, (short)0, c, false, false);
}

// 16 contiguous f16 (two b128 loads) -> B-frag chunk
__device__ __forceinline__ v16h load_contig16(const _Float16* p) {
  v8h lo = *(const v8h*)p;
  v8h hi = *(const v8h*)(p + 8);
  return __builtin_shufflevector(lo, hi, 0,1,2,3,4,5,6,7,8,9,10,11,12,13,14,15);
}
// two separated 8-element chunks -> A-frag (K at 8*half and 16+8*half)
__device__ __forceinline__ v16h load_split8(const _Float16* p0, const _Float16* p1) {
  v8h lo = *(const v8h*)p0;
  v8h hi = *(const v8h*)p1;
  return __builtin_shufflevector(lo, hi, 0,1,2,3,4,5,6,7,8,9,10,11,12,13,14,15);
}

// reductions within each 16-lane half (matches C-layout row split)
__device__ __forceinline__ float redmax16(float v) {
  #pragma unroll
  for (int m = 1; m < 16; m <<= 1) v = fmaxf(v, __shfl_xor(v, m, 32));
  return v;
}
__device__ __forceinline__ float redsum16(float v) {
  #pragma unroll
  for (int m = 1; m < 16; m <<= 1) v += __shfl_xor(v, m, 32);
  return v;
}

__global__ void zero_out_kernel(float* out) {
  if (threadIdx.x == 0 && blockIdx.x == 0) out[0] = 0.0f;
}

// f32 seq -> f16 copy, f16 transpose, row norms pre-scaled by log2(e)
__global__ void prep_kernel(const float* __restrict__ seq, _Float16* __restrict__ seqh,
                            _Float16* __restrict__ seqT, float* __restrict__ norms,
                            int T, int BT) {
  int r = blockIdx.x * blockDim.x + threadIdx.x;
  if (r >= BT) return;
  int b = r / T, t = r - b * T;
  const float* p = seq + (size_t)r * DFEAT;
  float n = 0.0f;
  #pragma unroll
  for (int d = 0; d < DFEAT; ++d) {
    float v = p[d];
    n += v * v;
    seqh[(size_t)r * DFEAT + d] = (_Float16)v;
    seqT[((size_t)b * DFEAT + d) * T + t] = (_Float16)v;
  }
  norms[r] = n * LOG2E;  // bias directly in log2 domain
}

// Stage 1: flash-attention.  snn[dir][i][:] = softmax_j(2 x_i.y_j - |y_j|^2) @ y
__global__ void __launch_bounds__(128)
stage1_snn(const _Float16* __restrict__ seqh, const _Float16* __restrict__ seqT,
           const float* __restrict__ norms, const int* __restrict__ src_len,
           const long long* __restrict__ comb, _Float16* __restrict__ snn, int T) {
  const int dir  = blockIdx.y;
  const int pair = dir >> 1;
  long long ia = comb[2 * pair + 0];
  long long ib = comb[2 * pair + 1];
  if (dir & 1) { long long tt = ia; ia = ib; ib = tt; }
  const int Ly = src_len[ib] >> 2;

  const int lane  = threadIdx.x & 31;
  const int wave  = threadIdx.x >> 5;
  const int lrow  = lane & 15;
  const int lhalf = lane >> 4;
  const int i0    = (blockIdx.x * 4 + wave) * 16;

  const _Float16* X  = seqh + (size_t)ia * T * DFEAT;
  const _Float16* Y  = seqh + (size_t)ib * T * DFEAT;
  const _Float16* YT = seqT + (size_t)ib * DFEAT * T;
  const float*    y2 = norms + (size_t)ib * T;  // already *log2e

  const _Float16* xr = X + (size_t)(i0 + lrow) * DFEAT + 16 * lhalf;
  v16h afrag = load_contig16(xr);

  v8f acc0 = {}; v8f acc1 = {};           // snn accumulators, D cols 0-15 / 16-31
  float m[8], l[8];
  #pragma unroll
  for (int g = 0; g < 8; ++g) { m[g] = -1e30f; l[g] = 0.0f; }

  __shared__ __align__(16) _Float16 ldsP[4][16 * 32];
  _Float16* P = ldsP[wave];

  const int ntiles = (Ly + 31) >> 5;
  for (int jt = 0; jt < ntiles; ++jt) {
    const int j0 = jt << 5;
    v16h b0 = load_contig16(Y + (size_t)(j0 + lrow) * DFEAT + 16 * lhalf);
    v16h b1 = load_contig16(Y + (size_t)(j0 + 16 + lrow) * DFEAT + 16 * lhalf);
    v8f zc = {};
    v8f s0 = wmma16x16x32(afrag, b0, zc);
    v8f s1 = wmma16x16x32(afrag, b1, zc);

    const int  c0 = j0 + lrow;      // always < T: in-bounds, load bias unconditionally
    const int  c1 = c0 + 16;
    const bool vv0 = c0 < Ly;
    const bool vv1 = c1 < Ly;
    const float bias0 = y2[c0];
    const float bias1 = y2[c1];

    float e0[8], e1[8], tmax[8];
    #pragma unroll
    for (int g = 0; g < 8; ++g) {
      e0[g] = vv0 ? (2.0f * LOG2E * s0[g] - bias0) : -1e30f;  // log2-domain logits
      e1[g] = vv1 ? (2.0f * LOG2E * s1[g] - bias1) : -1e30f;
      tmax[g] = fmaxf(e0[g], e1[g]);
    }
    #pragma unroll
    for (int g = 0; g < 8; ++g) tmax[g] = redmax16(tmax[g]);

    float scale[8], p0[8], p1[8], tsum[8];
    #pragma unroll
    for (int g = 0; g < 8; ++g) {
      float mn = fmaxf(m[g], tmax[g]);
      scale[g] = __builtin_amdgcn_exp2f(m[g] - mn);
      m[g] = mn;
      p0[g] = vv0 ? __builtin_amdgcn_exp2f(e0[g] - mn) : 0.0f;
      p1[g] = vv1 ? __builtin_amdgcn_exp2f(e1[g] - mn) : 0.0f;
      tsum[g] = p0[g] + p1[g];
    }
    #pragma unroll
    for (int g = 0; g < 8; ++g) tsum[g] = redsum16(tsum[g]);
    #pragma unroll
    for (int g = 0; g < 8; ++g) {
      l[g] = l[g] * scale[g] + tsum[g];
      acc0[g] *= scale[g];
      acc1[g] *= scale[g];
    }

    // transpose P (C-layout -> A-layout) through LDS; same-wave DS ops are in order
    #pragma unroll
    for (int g = 0; g < 8; ++g) {
      const int r = g + 8 * lhalf;
      P[r * 32 + lrow]      = (_Float16)p0[g];
      P[r * 32 + 16 + lrow] = (_Float16)p1[g];
    }
    const _Float16* pr = P + lrow * 32 + 16 * lhalf;
    v16h pfrag = load_contig16(pr);

    v16h bt0 = load_contig16(YT + (size_t)lrow * T        + j0 + 16 * lhalf);
    v16h bt1 = load_contig16(YT + (size_t)(16 + lrow) * T + j0 + 16 * lhalf);
    acc0 = wmma16x16x32(pfrag, bt0, acc0);
    acc1 = wmma16x16x32(pfrag, bt1, acc1);
  }

  _Float16* outp = snn + ((size_t)dir * T + i0) * DFEAT;
  #pragma unroll
  for (int g = 0; g < 8; ++g) {
    const int r = g + 8 * lhalf;
    const float inv = 1.0f / l[g];
    outp[r * DFEAT + lrow]      = (_Float16)(acc0[g] * inv);
    outp[r * DFEAT + 16 + lrow] = (_Float16)(acc1[g] * inv);
  }
}

// Stage 2: streaming softmax moments over k of 2 snn_i.x_k - |x_k|^2, Welford combine.
__global__ void __launch_bounds__(128)
stage2_loss(const _Float16* __restrict__ seqh, const float* __restrict__ norms,
            const int* __restrict__ src_len, const long long* __restrict__ comb,
            const _Float16* __restrict__ snn, float* __restrict__ out,
            int T, float invC) {
  const int dir  = blockIdx.y;
  const int pair = dir >> 1;
  long long ia = comb[2 * pair + 0];
  long long ib = comb[2 * pair + 1];
  if (dir & 1) { long long tt = ia; ia = ib; ib = tt; }
  const int Lx = src_len[ia] >> 2;

  const int lane  = threadIdx.x & 31;
  const int wave  = threadIdx.x >> 5;
  const int lrow  = lane & 15;
  const int lhalf = lane >> 4;
  const int i0    = (blockIdx.x * 4 + wave) * 16;

  const _Float16* S = snn + (size_t)dir * T * DFEAT;
  const _Float16* X = seqh + (size_t)ia * T * DFEAT;
  const float*   x2 = norms + (size_t)ia * T;  // already *log2e

  const _Float16* sr = S + (size_t)(i0 + lrow) * DFEAT + 16 * lhalf;
  v16h afrag = load_contig16(sr);

  // per-row state: running max (log2), weight W, weighted mean A (absolute k), M2 = sum w*(k-A)^2
  float m[8], W[8], A[8], M2[8];
  #pragma unroll
  for (int g = 0; g < 8; ++g) { m[g] = -1e30f; W[g] = 0.0f; A[g] = 0.0f; M2[g] = 0.0f; }

  // tile-local k offsets around tile center kc = k0+16: exact, small-magnitude
  const float d0 = (float)(lrow - 16);   // for cols k0..k0+15
  const float d1 = (float)lrow;          // for cols k0+16..k0+31
  const float d0sq = d0 * d0, d1sq = d1 * d1;

  const int ntiles = (Lx + 31) >> 5;
  for (int kt = 0; kt < ntiles; ++kt) {
    const int k0 = kt << 5;
    v16h b0 = load_contig16(X + (size_t)(k0 + lrow) * DFEAT + 16 * lhalf);
    v16h b1 = load_contig16(X + (size_t)(k0 + 16 + lrow) * DFEAT + 16 * lhalf);
    v8f zc = {};
    v8f s0 = wmma16x16x32(afrag, b0, zc);
    v8f s1 = wmma16x16x32(afrag, b1, zc);

    const int  c0 = k0 + lrow;
    const int  c1 = c0 + 16;
    const bool vv0 = c0 < Lx;
    const bool vv1 = c1 < Lx;
    const float bias0 = x2[c0];   // always in-bounds
    const float bias1 = x2[c1];

    float e0[8], e1[8], tmax[8];
    #pragma unroll
    for (int g = 0; g < 8; ++g) {
      e0[g] = vv0 ? (2.0f * LOG2E * s0[g] - bias0) : -1e30f;
      e1[g] = vv1 ? (2.0f * LOG2E * s1[g] - bias1) : -1e30f;
      tmax[g] = fmaxf(e0[g], e1[g]);
    }
    #pragma unroll
    for (int g = 0; g < 8; ++g) tmax[g] = redmax16(tmax[g]);

    float scale[8], ts[8], tsl[8], tsl2[8];
    #pragma unroll
    for (int g = 0; g < 8; ++g) {
      float mn = fmaxf(m[g], tmax[g]);
      scale[g] = __builtin_amdgcn_exp2f(m[g] - mn);
      m[g] = mn;
      float p0 = vv0 ? __builtin_amdgcn_exp2f(e0[g] - mn) : 0.0f;
      float p1 = vv1 ? __builtin_amdgcn_exp2f(e1[g] - mn) : 0.0f;
      ts[g]   = p0 + p1;
      tsl[g]  = p0 * d0 + p1 * d1;
      tsl2[g] = p0 * d0sq + p1 * d1sq;
    }
    #pragma unroll
    for (int g = 0; g < 8; ++g) ts[g]   = redsum16(ts[g]);
    #pragma unroll
    for (int g = 0; g < 8; ++g) tsl[g]  = redsum16(tsl[g]);
    #pragma unroll
    for (int g = 0; g < 8; ++g) tsl2[g] = redsum16(tsl2[g]);

    const float kc = (float)(k0 + 16);
    #pragma unroll
    for (int g = 0; g < 8; ++g) {
      const float w_t = ts[g];
      const float safe_w = fmaxf(w_t, 1e-30f);
      const float mean_t = kc + tsl[g] / safe_w;                 // tile mean (absolute k)
      const float m2_t   = tsl2[g] - tsl[g] * (tsl[g] / safe_w); // tile sum w*(k-mean_t)^2 >= 0
      const float Wold = W[g] * scale[g];                        // rescaled running weight
      const float Wn   = Wold + w_t;
      const float invWn = (Wn > 0.0f) ? (1.0f / Wn) : 0.0f;
      const float delta = mean_t - A[g];
      A[g]  = A[g] + delta * (w_t * invWn);
      M2[g] = M2[g] * scale[g] + m2_t + delta * delta * (Wold * w_t * invWn);
      W[g]  = Wn;
    }
  }

  if (lrow == 0) {  // lanes 0 (rows 0..7) and 16 (rows 8..15)
    float part = 0.0f;
    #pragma unroll
    for (int g = 0; g < 8; ++g) {
      const int i = i0 + g + 8 * lhalf;
      if (i < Lx) {
        float var = M2[g] / W[g];
        var = fmaxf(var, 1e-20f);
        const float di = (float)i - A[g];
        part += di * di / var + 0.005f * __logf(var);  // penalty*log(sqrt(var))
      }
    }
    atomicAdd(out, part * invC);
  }
}

extern "C" void kernel_launch(void* const* d_in, const int* in_sizes, int n_in,
                              void* d_out, int out_size, void* d_ws, size_t ws_size,
                              hipStream_t stream) {
  const float*     seq     = (const float*)d_in[0];
  const int*       src_len = (const int*)d_in[1];
  const long long* comb    = (const long long*)d_in[2];  // int64 pairs

  const int B    = in_sizes[1];               // 8
  const int C    = in_sizes[2] / 2;           // 8
  const int T    = in_sizes[0] / (B * DFEAT); // 4096
  const int BT   = B * T;
  const int ndir = 2 * C;                     // 16

  char* ws = (char*)d_ws;
  size_t off = 0;
  auto wsalloc = [&](size_t bytes) -> char* {
    char* p = ws + off;
    off = (off + bytes + 255) & ~(size_t)255;
    return p;
  };
  _Float16* seqh = (_Float16*)wsalloc((size_t)BT * DFEAT * sizeof(_Float16));
  _Float16* seqT = (_Float16*)wsalloc((size_t)BT * DFEAT * sizeof(_Float16));
  float*    nrm  = (float*)wsalloc((size_t)BT * sizeof(float));
  _Float16* snn  = (_Float16*)wsalloc((size_t)ndir * T * DFEAT * sizeof(_Float16));

  zero_out_kernel<<<1, 1, 0, stream>>>((float*)d_out);
  prep_kernel<<<(BT + 127) / 128, 128, 0, stream>>>(seq, seqh, seqT, nrm, T, BT);
  dim3 grid(T / 64, ndir);
  stage1_snn<<<grid, 128, 0, stream>>>(seqh, seqT, nrm, src_len, comb, snn, T);
  stage2_loss<<<grid, 128, 0, stream>>>(seqh, nrm, src_len, comb, snn, (float*)d_out,
                                        T, 1.0f / (float)C);
}
